// Translution_25288767439392
// MI455X (gfx1250) — matrix operand order repaired and checked by reading, not compiled, not run on hardware
//
#include <hip/hip_runtime.h>
#include <stdint.h>

#define N_SEQ 512
#define DIM   256
#define HEADS 8
#define LDW   264          // padded LDS row stride (bf16 elements): 264*2B = 16B-aligned rows
#define LN_EPS 1e-5f

typedef __attribute__((ext_vector_type(16))) __bf16 v16bf;
typedef __attribute__((ext_vector_type(8)))  float  v8f;

union V16 {
    v16bf v;
    uint4 q[2];
    unsigned short u[16];
};

__device__ __forceinline__ unsigned short f2bf(float f) {
    union { float f; unsigned int u; } a; a.f = f;
    unsigned int u = a.u;
    unsigned int r = u + 0x7FFFu + ((u >> 16) & 1u);   // round-to-nearest-even
    return (unsigned short)(r >> 16);
}

__device__ __forceinline__ v8f wmma_bf16(v16bf a, v16bf b, v8f c) {
    // D = A(16x32 bf16) * B(32x16 bf16) + C(16x16 f32)
    return __builtin_amdgcn_wmma_f32_16x16x32_bf16(
        false, a, false, b, (short)0, c, false, false);
}

__device__ __forceinline__ void atomic_add_f32(float* p, float v) {
    __hip_atomic_fetch_add(p, v, __ATOMIC_RELAXED, __HIP_MEMORY_SCOPE_AGENT);
}

// ---------------------------------------------------------------------------
// Pass 0: LayerNorm, emit xn as bf16 (A-operand source for all GEMMs)
// ---------------------------------------------------------------------------
__global__ __launch_bounds__(32) void ln_kernel(
    const float* __restrict__ x, const float* __restrict__ gamma,
    const float* __restrict__ beta, unsigned short* __restrict__ xnbf) {
    int row  = blockIdx.x;
    int lane = threadIdx.x;
    const float* xr = x + (size_t)row * DIM;
    float vals[8];
    float s = 0.f;
#pragma unroll
    for (int i = 0; i < 8; ++i) { vals[i] = xr[lane * 8 + i]; s += vals[i]; }
#pragma unroll
    for (int off = 16; off; off >>= 1) s += __shfl_xor(s, off, 32);
    float mu = s * (1.0f / DIM);
    float vs = 0.f;
#pragma unroll
    for (int i = 0; i < 8; ++i) { float dlt = vals[i] - mu; vs += dlt * dlt; }
#pragma unroll
    for (int off = 16; off; off >>= 1) vs += __shfl_xor(vs, off, 32);
    float inv = rsqrtf(vs * (1.0f / DIM) + LN_EPS);
    unsigned int packed[4];
#pragma unroll
    for (int i = 0; i < 4; ++i) {
        int c0 = lane * 8 + 2 * i, c1 = c0 + 1;
        float y0 = (vals[2 * i]     - mu) * inv * gamma[c0] + beta[c0];
        float y1 = (vals[2 * i + 1] - mu) * inv * gamma[c1] + beta[c1];
        packed[i] = (unsigned int)f2bf(y0) | ((unsigned int)f2bf(y1) << 16);
    }
    uint4 st; st.x = packed[0]; st.y = packed[1]; st.z = packed[2]; st.w = packed[3];
    *(uint4*)(xnbf + (size_t)row * DIM + lane * 8) = st;
}

__global__ void zero_kernel(float* __restrict__ p, int n) {
    int i = blockIdx.x * blockDim.x + threadIdx.x;
    if (i < n) p[i] = 0.f;
}

// Cooperative transposed bf16 weight load: W[c][e] (global f32) -> lds[e][c]
__device__ __forceinline__ void load_wT(const float* __restrict__ W,
                                        unsigned short* __restrict__ lds,
                                        int tid, int nthreads) {
    for (int idx = tid; idx < DIM * DIM; idx += nthreads) {
        int c = idx >> 8, e = idx & 255;            // coalesced over e
        lds[e * LDW + c] = f2bf(W[idx]);
    }
}

// ---------------------------------------------------------------------------
// Pass 1: per-diagonal scores.  Workgroup = one d.  LDS holds WqT[d], WkT[511-d].
// S[n][h][d] = <Wq[d]^T xn[n-d], Wk[511-d]^T xn[n]>_head * scale
// 256 threads = 8 waves (2 per SIMD even at 1 WG/WGP due to 270KB LDS)
// ---------------------------------------------------------------------------
__global__ __launch_bounds__(256) void scores_kernel(
    const unsigned short* __restrict__ xnbf,
    const float* __restrict__ Wq, const float* __restrict__ Wk,
    float* __restrict__ S) {
    extern __shared__ unsigned short smem[];
    unsigned short* wqT = smem;
    unsigned short* wkT = smem + 256 * LDW;

    int d   = blockIdx.x;
    int tid = threadIdx.x;
    load_wT(Wq + (size_t)d * DIM * DIM,               wqT, tid, 256);
    load_wT(Wk + (size_t)(N_SEQ - 1 - d) * DIM * DIM, wkT, tid, 256);
    __syncthreads();

    int wave = tid >> 5, lane = tid & 31;
    int hi = lane >> 4, col = lane & 15;
    int R = N_SEQ - d;                                 // valid diagonal length
    const float scale = 0.17677669529663687f;          // 1/sqrt(32)

    for (int t = wave * 16; t < R; t += 128) {
        int mrow = t + col; if (mrow > R - 1) mrow = R - 1;
        int nrow = mrow + d;
        V16 aq[8], ak[8];
#pragma unroll
        for (int k8 = 0; k8 < 8; ++k8) {               // A layout: K 0-7/16-23 (lo lanes), 8-15/24-31 (hi)
            int kc = k8 * 32 + hi * 8;
            const unsigned short* qm = xnbf + (size_t)mrow * DIM + kc;
            const unsigned short* qn = xnbf + (size_t)nrow * DIM + kc;
            aq[k8].q[0] = *(const uint4*)(qm);
            aq[k8].q[1] = *(const uint4*)(qm + 16);
            ak[k8].q[0] = *(const uint4*)(qn);
            ak[k8].q[1] = *(const uint4*)(qn + 16);
        }
        for (int h = 0; h < HEADS; ++h) {
            float hacc[8] = {0, 0, 0, 0, 0, 0, 0, 0};
#pragma unroll
            for (int sub = 0; sub < 2; ++sub) {        // head = two 16-col eTiles
                int eb = h * 32 + sub * 16 + col;
                const unsigned short* bq = wqT + eb * LDW + hi * 8;
                const unsigned short* bk = wkT + eb * LDW + hi * 8;
                v8f cq = {0.f, 0.f, 0.f, 0.f, 0.f, 0.f, 0.f, 0.f};
                v8f ck = {0.f, 0.f, 0.f, 0.f, 0.f, 0.f, 0.f, 0.f};
#pragma unroll
                for (int k8 = 0; k8 < 8; ++k8) {
                    V16 b0, b1;
                    b0.q[0] = *(const uint4*)(bq + k8 * 32);
                    b0.q[1] = *(const uint4*)(bq + k8 * 32 + 16);
                    b1.q[0] = *(const uint4*)(bk + k8 * 32);
                    b1.q[1] = *(const uint4*)(bk + k8 * 32 + 16);
                    cq = wmma_bf16(aq[k8].v, b0.v, cq);
                    ck = wmma_bf16(ak[k8].v, b1.v, ck);
                }
#pragma unroll
                for (int v = 0; v < 8; ++v) {          // C layout: (M=v+8*hi, N=col)
                    float p = cq[v] * ck[v];
                    p += __shfl_xor(p, 1, 16);
                    p += __shfl_xor(p, 2, 16);
                    p += __shfl_xor(p, 4, 16);
                    p += __shfl_xor(p, 8, 16);
                    hacc[v] += p;
                }
            }
            if (col == 0) {
#pragma unroll
                for (int v = 0; v < 8; ++v) {
                    int m = t + v + 8 * hi;
                    if (m < R) {
                        int n = m + d;
                        S[((size_t)n * HEADS + h) * N_SEQ + d] = hacc[v] * scale;
                    }
                }
            }
        }
    }
}

// ---------------------------------------------------------------------------
// Pass 2: causal softmax over d in [0, n] per (n, h) row (in place)
// 4 waves per block, one (n,h) row per wave
// ---------------------------------------------------------------------------
__global__ __launch_bounds__(128) void softmax_kernel(float* __restrict__ S) {
    int row = blockIdx.x * 4 + (threadIdx.x >> 5);     // n*8 + h
    int n   = row >> 3;
    int len = n + 1;
    float* p = S + (size_t)row * N_SEQ;
    int lane = threadIdx.x & 31;
    float mx = -3.4e38f;
    for (int i = lane; i < len; i += 32) mx = fmaxf(mx, p[i]);
#pragma unroll
    for (int off = 16; off; off >>= 1) mx = fmaxf(mx, __shfl_xor(mx, off, 32));
    float sum = 0.f;
    for (int i = lane; i < len; i += 32) { float e = __expf(p[i] - mx); p[i] = e; sum += e; }
#pragma unroll
    for (int off = 16; off; off >>= 1) sum += __shfl_xor(sum, off, 32);
    float inv = 1.0f / sum;
    for (int i = lane; i < len; i += 32) p[i] *= inv;
}

// ---------------------------------------------------------------------------
// Pass 3: per-diagonal AV.  V(d,m) = xn[m] @ Wv[d]; out_inner[n] += P[n,h,d] * V
// 135KB LDS -> 2 WGs/WGP; 256 threads each
// ---------------------------------------------------------------------------
__global__ __launch_bounds__(256) void av_kernel(
    const unsigned short* __restrict__ xnbf, const float* __restrict__ Wv,
    const float* __restrict__ S, float* __restrict__ out_inner) {
    extern __shared__ unsigned short smemv[];
    unsigned short* wvT = smemv;

    int d   = blockIdx.x;
    int tid = threadIdx.x;
    load_wT(Wv + (size_t)d * DIM * DIM, wvT, tid, 256);
    __syncthreads();

    int wave = tid >> 5, lane = tid & 31;
    int hi = lane >> 4, col = lane & 15;
    int R = N_SEQ - d;

    for (int t = wave * 16; t < R; t += 128) {
        int mrow = t + col; if (mrow > R - 1) mrow = R - 1;
        V16 a[8];
#pragma unroll
        for (int k8 = 0; k8 < 8; ++k8) {
            int kc = k8 * 32 + hi * 8;
            const unsigned short* qm = xnbf + (size_t)mrow * DIM + kc;
            a[k8].q[0] = *(const uint4*)(qm);
            a[k8].q[1] = *(const uint4*)(qm + 16);
        }
        for (int h = 0; h < HEADS; ++h) {
            float pr[8];
#pragma unroll
            for (int v = 0; v < 8; ++v) {
                int m = t + v + 8 * hi;
                pr[v] = (m < R) ? S[((size_t)(m + d) * HEADS + h) * N_SEQ + d] : 0.f;
            }
#pragma unroll
            for (int sub = 0; sub < 2; ++sub) {
                int eb = h * 32 + sub * 16 + col;
                const unsigned short* bv = wvT + eb * LDW + hi * 8;
                v8f c = {0.f, 0.f, 0.f, 0.f, 0.f, 0.f, 0.f, 0.f};
#pragma unroll
                for (int k8 = 0; k8 < 8; ++k8) {
                    V16 b;
                    b.q[0] = *(const uint4*)(bv + k8 * 32);
                    b.q[1] = *(const uint4*)(bv + k8 * 32 + 16);
                    c = wmma_bf16(a[k8].v, b.v, c);
                }
#pragma unroll
                for (int v = 0; v < 8; ++v) {
                    int m = t + v + 8 * hi;
                    if (m < R)
                        atomic_add_f32(&out_inner[(size_t)(m + d) * DIM + eb], c[v] * pr[v]);
                }
            }
        }
    }
}

// ---------------------------------------------------------------------------
// Pass 4: out = out_inner @ Wo + bo   (512x256 @ 256x256)
// ---------------------------------------------------------------------------
__global__ __launch_bounds__(128) void proj_kernel(
    const float* __restrict__ out_inner, const float* __restrict__ Wo,
    const float* __restrict__ bo, float* __restrict__ out) {
    extern __shared__ unsigned short smemo[];
    unsigned short* woT = smemo;

    int tid = threadIdx.x;
    load_wT(Wo, woT, tid, 128);
    __syncthreads();

    int wave = tid >> 5, lane = tid & 31;
    int hi = lane >> 4, col = lane & 15;
    int rowBase = blockIdx.x * 16;
    int r = rowBase + col;
    const float* arow = out_inner + (size_t)r * DIM;

    V16 a[8];
#pragma unroll
    for (int k8 = 0; k8 < 8; ++k8) {
        int kc = k8 * 32 + hi * 8;
        float4 f0 = *(const float4*)(arow + kc);
        float4 f1 = *(const float4*)(arow + kc + 4);
        float4 f2 = *(const float4*)(arow + kc + 16);
        float4 f3 = *(const float4*)(arow + kc + 20);
        a[k8].u[0]  = f2bf(f0.x); a[k8].u[1]  = f2bf(f0.y);
        a[k8].u[2]  = f2bf(f0.z); a[k8].u[3]  = f2bf(f0.w);
        a[k8].u[4]  = f2bf(f1.x); a[k8].u[5]  = f2bf(f1.y);
        a[k8].u[6]  = f2bf(f1.z); a[k8].u[7]  = f2bf(f1.w);
        a[k8].u[8]  = f2bf(f2.x); a[k8].u[9]  = f2bf(f2.y);
        a[k8].u[10] = f2bf(f2.z); a[k8].u[11] = f2bf(f2.w);
        a[k8].u[12] = f2bf(f3.x); a[k8].u[13] = f2bf(f3.y);
        a[k8].u[14] = f2bf(f3.z); a[k8].u[15] = f2bf(f3.w);
    }
    for (int et = wave; et < 16; et += 4) {
        int eb = et * 16 + col;
        const unsigned short* bw = woT + eb * LDW + hi * 8;
        v8f c = {0.f, 0.f, 0.f, 0.f, 0.f, 0.f, 0.f, 0.f};
#pragma unroll
        for (int k8 = 0; k8 < 8; ++k8) {
            V16 b;
            b.q[0] = *(const uint4*)(bw + k8 * 32);
            b.q[1] = *(const uint4*)(bw + k8 * 32 + 16);
            c = wmma_bf16(a[k8].v, b.v, c);
        }
        float bias = bo[eb];
#pragma unroll
        for (int v = 0; v < 8; ++v) {
            int rr = rowBase + v + 8 * hi;
            out[(size_t)rr * DIM + eb] = c[v] + bias;
        }
    }
}

// ---------------------------------------------------------------------------
extern "C" void kernel_launch(void* const* d_in, const int* in_sizes, int n_in,
                              void* d_out, int out_size, void* d_ws, size_t ws_size,
                              hipStream_t stream) {
    const float* x     = (const float*)d_in[0];
    const float* Wq    = (const float*)d_in[1];
    const float* Wk    = (const float*)d_in[2];
    const float* Wv    = (const float*)d_in[3];
    const float* gamma = (const float*)d_in[4];
    const float* beta  = (const float*)d_in[5];
    const float* Wo    = (const float*)d_in[6];
    const float* bo    = (const float*)d_in[7];
    float* out = (float*)d_out;

    char* ws = (char*)d_ws;
    unsigned short* xnbf   = (unsigned short*)ws;                       // 512*256*2   = 262144 B
    float* S               = (float*)(ws + 262144);                     // 512*8*512*4 = 8388608 B
    float* out_inner       = (float*)(ws + 262144 + 8388608);           // 512*256*4   = 524288 B

    ln_kernel<<<N_SEQ, 32, 0, stream>>>(x, gamma, beta, xnbf);
    zero_kernel<<<512, 256, 0, stream>>>(out_inner, N_SEQ * DIM);

    size_t lds2 = (size_t)2 * 256 * LDW * sizeof(unsigned short);       // 270336 B (<320KB WGP LDS)
    size_t lds1 = (size_t)256 * LDW * sizeof(unsigned short);           // 135168 B
    scores_kernel<<<N_SEQ, 256, lds2, stream>>>(xnbf, Wq, Wk, S);
    softmax_kernel<<<N_SEQ * HEADS / 4, 128, 0, stream>>>(S);
    av_kernel<<<N_SEQ, 256, lds1, stream>>>(xnbf, Wv, S, out_inner);
    proj_kernel<<<N_SEQ / 16, 128, lds1, stream>>>(out_inner, Wo, bo, out);
}
